// NConvUNet_59416577572981
// MI455X (gfx1250) — matrix operand order repaired and checked by reading, not compiled, not run on hardware
//
#include <hip/hip_runtime.h>
#include <math.h>

typedef __attribute__((ext_vector_type(16))) _Float16 v16h;
typedef __attribute__((ext_vector_type(8)))  float    v8f;

#define EPSN 1e-20f

__device__ __forceinline__ float softplusf(float x) {
    return (x > 20.0f) ? x : log1pf(expf(x));
}

// ---------------------------------------------------------------------------
// Weight prep: softplus(w_raw) packed directly into the WMMA B-fragment layout
// B is 32x16 (KxN) per k-step: lane L -> n = L&15, k-half = L>>4;
// element e of lane L -> k = kstep*32 + (L>>4)*16 + e.
// frag[((ks*32 + lane)*16) + e]  (one v16h per lane, contiguous & coalesced)
// Also emits per-out-channel kernel sums s[n] for the cout = denom/s path.
// ---------------------------------------------------------------------------
__global__ void prep_frag(const float* __restrict__ w, int Cin, int KK,
                          int KSTEPS, _Float16* __restrict__ frag,
                          float* __restrict__ sums) {
    int total = KSTEPS * 32 * 16;
    int K = Cin * KK;
    for (int i = blockIdx.x * blockDim.x + threadIdx.x; i < total;
         i += gridDim.x * blockDim.x) {
        int e    = i & 15;
        int lane = (i >> 4) & 31;
        int ks   = i >> 9;
        int kg   = ks * 32 + ((lane >> 4) << 4) + e;
        int n    = lane & 15;
        float v = 0.0f;
        if (n < 8 && kg < K) {
            int ci = kg / KK, kk = kg % KK;
            v = softplusf(w[(n * Cin + ci) * KK + kk]);
        }
        frag[i] = (_Float16)v;
    }
    int t = blockIdx.x * blockDim.x + threadIdx.x;
    if (t < 8) {
        float s = 0.0f;
        for (int j = 0; j < K; ++j) s += softplusf(w[t * K + j]);
        sums[t] = s;
    }
}

__global__ void prep_w7(const float* __restrict__ w7, float* __restrict__ w7sp,
                        float* __restrict__ s7) {
    if (threadIdx.x == 0 && blockIdx.x == 0) {
        float s = 0.0f;
        for (int i = 0; i < 8; ++i) {
            float v = softplusf(w7[i]);
            w7sp[i] = v;
            s += v;
        }
        s7[0] = s;
    }
}

// ---------------------------------------------------------------------------
// WMMA normalized conv. One wave (32 threads) per workgroup computes 8 output
// pixels x 8 output channels for one (b, y, x-tile).
// M rows (16): m = pixel*2 + sel, sel 0 -> x*c (nomin), sel 1 -> c (denom)
// (nomin & denom share the weight matrix B, so they share every WMMA).
// Input = concat(A, B-src) along channels; each source may be 2x-upsampled
// (nearest) on the fly -> decoder concat+up2 folded into the loader.
// B fragments are staged to LDS once per block via global_load_async_to_lds.
// ---------------------------------------------------------------------------
template <int KK, int KH, int KW, int PAD, int KSTEPS>
__global__ __launch_bounds__(32) void nconv_wmma(
    const float* __restrict__ xA, const float* __restrict__ cA, int CinA, int upA,
    const float* __restrict__ xB, const float* __restrict__ cB, int CinB, int upB,
    const _Float16* __restrict__ wfrag, const float* __restrict__ sums,
    const float* __restrict__ bias, int H, int W,
    float* __restrict__ xOut, float* __restrict__ cOut) {

    __shared__ _Float16 sXC[16 * 5 * 12];                 // x*c window, f16
    __shared__ _Float16 sC [16 * 5 * 12];                 // c   window, f16
    __shared__ __align__(32) _Float16 sB[KSTEPS * 512];   // B fragments
    __shared__ unsigned short sMap[KSTEPS * 32];          // k->(ci,dy,dx), 0xFFFF=pad

    const int lane = (int)threadIdx.x;
    const int b = blockIdx.z, y = blockIdx.y, x0 = blockIdx.x * 8;
    const int Cin = CinA + CinB;
    constexpr int WIN = 8 + KW - 1;

    // --- async-copy the weight fragments (verbatim bytes) into LDS ---------
    {
        const unsigned lbase = (unsigned)(unsigned long long)(const void*)&sB[0];
        const unsigned long long gbase = (unsigned long long)(const void*)wfrag;
        // KSTEPS*1024 bytes total, 16B chunks, 32 lanes -> uniform trip count
        for (int i = lane; i < KSTEPS * 64; i += 32) {
            unsigned lofs = lbase + (unsigned)(i * 16);
            unsigned long long ga = gbase + (unsigned long long)(i * 16);
            asm volatile("global_load_async_to_lds_b128 %0, %1, off"
                         :: "v"(lofs), "v"(ga) : "memory");
        }
    }

    // --- k-index decode table ----------------------------------------------
    for (int k = lane; k < KSTEPS * 32; k += 32) {
        unsigned short mc = 0xFFFFu;
        if (k < Cin * KK) {
            int ci = k / KK, kk = k % KK;     // division by constant
            int dy = kk / KW, dx = kk % KW;
            mc = (unsigned short)((ci << 6) | (dy << 3) | dx);
        }
        sMap[k] = mc;
    }

    // --- stage input window into LDS (zero-padded, f16, x*c fused) ---------
    const int tot = Cin * KH * WIN;
    for (int idx = lane; idx < tot; idx += 32) {
        int col = idx % WIN;
        int rem = idx / WIN;
        int dy = rem % KH, ci = rem / KH;
        int yy = y + dy - PAD, xx = x0 + col - PAD;
        float xv = 0.0f, cv = 0.0f;
        if (yy >= 0 && yy < H && xx >= 0 && xx < W) {
            if (ci < CinA) {
                int Hs = upA ? (H >> 1) : H, Ws = upA ? (W >> 1) : W;
                int ys = upA ? (yy >> 1) : yy, xs = upA ? (xx >> 1) : xx;
                long o = (((long)(b * CinA + ci)) * Hs + ys) * Ws + xs;
                xv = xA[o]; cv = cA[o];
            } else {
                int cj = ci - CinA;
                int Hs = upB ? (H >> 1) : H, Ws = upB ? (W >> 1) : W;
                int ys = upB ? (yy >> 1) : yy, xs = upB ? (xx >> 1) : xx;
                long o = (((long)(b * CinB + cj)) * Hs + ys) * Ws + xs;
                xv = xB[o]; cv = cB[o];
            }
        }
        sXC[(ci * 5 + dy) * 12 + col] = (_Float16)(xv * cv);
        sC [(ci * 5 + dy) * 12 + col] = (_Float16)cv;
    }
    asm volatile("s_wait_asynccnt 0x0" ::: "memory");   // B frags landed in LDS
    __syncthreads();

    // --- K loop: A from window LDS, B from frag LDS, WMMA accumulate -------
    // A layout (16-bit A 16x32): lane&15 = M row; e0..15 -> k=(lane>>4)*16+e
    // (contiguous, matches B frag).
    const int m  = lane & 15;
    const int pi = m >> 1, sel = m & 1;
    const _Float16* src = sel ? sC : sXC;
    const int h8 = (lane >> 4) << 4;

    union { v8f v; float f[8]; } acc = {};
    const v16h* bv = ((const v16h*)sB) + lane;

#pragma unroll
    for (int ks = 0; ks < KSTEPS; ++ks) {
        union { v16h v; _Float16 h[16]; } af;
        const int kbase = ks * 32 + h8;
#pragma unroll
        for (int e = 0; e < 16; ++e) {
            int k = kbase + e;
            unsigned short mc = sMap[k];
            _Float16 v = (_Float16)0.0f;
            if (mc != 0xFFFFu) {
                int ci = mc >> 6, dy = (mc >> 3) & 7, dx = mc & 7;
                v = src[(ci * 5 + dy) * 12 + pi + dx];
            }
            af.h[e] = v;
        }
        v16h bf = bv[ks * 32];
        acc.v = __builtin_amdgcn_wmma_f32_16x16x32_f16(
            false, af.v, false, bf, (short)0, acc.v, false, false);
    }

    // --- epilogue: D layout -> lane&15 = N, rows r -> M = r + 8*(lane>>4) --
    const int n = lane & 15;
    const int h = lane >> 4;
    if (n < 8) {
        const float bn = bias[n], sn = sums[n];
        const long obase = (((long)(b * 8 + n)) * H + y) * W + x0;
#pragma unroll
        for (int r = 0; r < 8; r += 2) {
            int prow = (r + (h << 3)) >> 1;            // pixel 0..7
            float nom = acc.f[r], den = acc.f[r + 1];
            xOut[obase + prow] = nom / (den + EPSN) + bn;
            cOut[obase + prow] = den / sn;
        }
    }
}

// ---------------------------------------------------------------------------
// Confidence-argmax 2x2 pool: gather x at argmax(c), c_out = max(c)/4.
// argmax tie-break = first index, window order (0,0),(0,1),(1,0),(1,1).
// ---------------------------------------------------------------------------
__global__ void pool_gather(const float* __restrict__ x, const float* __restrict__ c,
                            float* __restrict__ xo, float* __restrict__ co,
                            int B, int C, int Ho, int Wo) {
    long total = (long)B * C * Ho * Wo;
    long i = (long)blockIdx.x * blockDim.x + threadIdx.x;
    if (i >= total) return;
    int wo = (int)(i % Wo);
    int ho = (int)((i / Wo) % Ho);
    int ch = (int)((i / ((long)Wo * Ho)) % C);
    int b  = (int)(i / ((long)Wo * Ho * C));
    int H = Ho * 2, W = Wo * 2;
    long base = (((long)(b * C + ch)) * H + 2 * ho) * W + 2 * wo;
    float c00 = c[base], c01 = c[base + 1], c10 = c[base + W], c11 = c[base + W + 1];
    int idx = 0; float best = c00;
    if (c01 > best) { best = c01; idx = 1; }
    if (c10 > best) { best = c10; idx = 2; }
    if (c11 > best) { best = c11; idx = 3; }
    long off = (long)(idx >> 1) * W + (idx & 1);
    xo[i] = x[base + off];
    co[i] = best * 0.25f;
}

// ---------------------------------------------------------------------------
// Final 1x1 nconv (Cin=8 -> Cout=1), pure pointwise, memory-bound.
// ---------------------------------------------------------------------------
__global__ void nconv1x1(const float* __restrict__ x, const float* __restrict__ c,
                         const float* __restrict__ wsp, const float* __restrict__ bias,
                         const float* __restrict__ s7,
                         float* __restrict__ xout, float* __restrict__ cout,
                         int B, int H, int W) {
    long sp = (long)H * W;
    long total = (long)B * sp;
    long i = (long)blockIdx.x * blockDim.x + threadIdx.x;
    if (i >= total) return;
    long b = i / sp, p = i % sp;
    const float* xb = x + b * 8 * sp + p;
    const float* cb = c + b * 8 * sp + p;
    float nom = 0.0f, den = 0.0f;
#pragma unroll
    for (int ci = 0; ci < 8; ++ci) {
        float w = wsp[ci];
        float cc = cb[ci * sp];
        nom += w * xb[ci * sp] * cc;
        den += w * cc;
    }
    xout[i] = nom / (den + EPSN) + bias[0];
    cout[i] = den / s7[0];
}

// ---------------------------------------------------------------------------

extern "C" void kernel_launch(void* const* d_in, const int* in_sizes, int n_in,
                              void* d_out, int out_size, void* d_ws, size_t ws_size,
                              hipStream_t stream) {
    (void)in_sizes; (void)n_in; (void)out_size; (void)ws_size;

    const float* x0 = (const float*)d_in[0];
    const float* c0 = (const float*)d_in[1];
    const float* w1 = (const float*)d_in[2];  const float* b1 = (const float*)d_in[3];
    const float* w2 = (const float*)d_in[4];  const float* b2 = (const float*)d_in[5];
    const float* w3 = (const float*)d_in[6];  const float* b3 = (const float*)d_in[7];
    const float* w4 = (const float*)d_in[8];  const float* b4 = (const float*)d_in[9];
    const float* w5 = (const float*)d_in[10]; const float* b5 = (const float*)d_in[11];
    const float* w6 = (const float*)d_in[12]; const float* b6 = (const float*)d_in[13];
    const float* w7 = (const float*)d_in[14]; const float* b7 = (const float*)d_in[15];

    const int B = 8, H = 352, W = 1216;
    const int H1 = H / 2, W1 = W / 2;      // 176 x 608
    const int H2 = H / 4, W2 = W / 4;      // 88 x 304
    const int H3 = H / 8, W3 = W / 8;      // 44 x 152

    const long S0 = (long)B * 8 * H * W;   // 27,394,048
    const long S1 = S0 / 4, S2 = S0 / 16, S3 = S0 / 64;

    float* ws = (float*)d_ws;
    long off = 0;
    float* Px  = ws + off; off += S0;  float* Pc  = ws + off; off += S0;
    float* Qx  = ws + off; off += S0;  float* Qc  = ws + off; off += S0;
    float* A1x = ws + off; off += S1;  float* A1c = ws + off; off += S1;
    float* B1x = ws + off; off += S1;  float* B1c = ws + off; off += S1;
    float* A2x = ws + off; off += S2;  float* A2c = ws + off; off += S2;
    float* B2x = ws + off; off += S2;  float* B2c = ws + off; off += S2;
    float* A3x = ws + off; off += S3;  float* A3c = ws + off; off += S3;
    float* B3x = ws + off; off += S3;  float* B3c = ws + off; off += S3;

    _Float16* fr = (_Float16*)(ws + off);           // frag region (32B aligned)
    _Float16* f1 = fr;               // KSTEPS=1 -> 512 halfs
    _Float16* f2 = f1 + 1 * 512;     // KSTEPS=7 -> 3584
    _Float16* f3 = f2 + 7 * 512;
    _Float16* f4 = f3 + 7 * 512;     // KSTEPS=5 -> 2560
    _Float16* f5 = f4 + 5 * 512;
    _Float16* f6 = f5 + 5 * 512;
    float* sm = (float*)(f6 + 5 * 512);
    float* s1 = sm;      float* s2 = sm + 8;  float* s3 = sm + 16;
    float* s4 = sm + 24; float* s5 = sm + 32; float* s6 = sm + 40;
    float* w7sp = sm + 48; float* s7 = sm + 56;

    // -- weight prep --------------------------------------------------------
    prep_frag<<<8, 256, 0, stream>>>(w1, 1,  25, 1, f1, s1);
    prep_frag<<<8, 256, 0, stream>>>(w2, 8,  25, 7, f2, s2);
    prep_frag<<<8, 256, 0, stream>>>(w3, 8,  25, 7, f3, s3);
    prep_frag<<<8, 256, 0, stream>>>(w4, 16, 9,  5, f4, s4);
    prep_frag<<<8, 256, 0, stream>>>(w5, 16, 9,  5, f5, s5);
    prep_frag<<<8, 256, 0, stream>>>(w6, 16, 9,  5, f6, s6);
    prep_w7  <<<1, 32,  0, stream>>>(w7, w7sp, s7);

    // -- encoder ------------------------------------------------------------
    dim3 g0(W / 8, H, B), g1(W1 / 8, H1, B), g2(W2 / 8, H2, B), g3(W3 / 8, H3, B);
    nconv_wmma<25,5,5,2,1><<<g0, 32, 0, stream>>>(x0, c0, 1, 0, nullptr, nullptr, 0, 0,
                                                  f1, s1, b1, H, W, Px, Pc);
    nconv_wmma<25,5,5,2,7><<<g0, 32, 0, stream>>>(Px, Pc, 8, 0, nullptr, nullptr, 0, 0,
                                                  f2, s2, b2, H, W, Qx, Qc);
    nconv_wmma<25,5,5,2,7><<<g0, 32, 0, stream>>>(Qx, Qc, 8, 0, nullptr, nullptr, 0, 0,
                                                  f3, s3, b3, H, W, Px, Pc);
    // level 1
    pool_gather<<<(unsigned)((S1 + 255) / 256), 256, 0, stream>>>(Px, Pc, A1x, A1c, B, 8, H1, W1);
    nconv_wmma<25,5,5,2,7><<<g1, 32, 0, stream>>>(A1x, A1c, 8, 0, nullptr, nullptr, 0, 0,
                                                  f2, s2, b2, H1, W1, B1x, B1c);
    nconv_wmma<25,5,5,2,7><<<g1, 32, 0, stream>>>(B1x, B1c, 8, 0, nullptr, nullptr, 0, 0,
                                                  f3, s3, b3, H1, W1, A1x, A1c);
    // level 2
    pool_gather<<<(unsigned)((S2 + 255) / 256), 256, 0, stream>>>(A1x, A1c, A2x, A2c, B, 8, H2, W2);
    nconv_wmma<25,5,5,2,7><<<g2, 32, 0, stream>>>(A2x, A2c, 8, 0, nullptr, nullptr, 0, 0,
                                                  f2, s2, b2, H2, W2, B2x, B2c);
    // level 3
    pool_gather<<<(unsigned)((S3 + 255) / 256), 256, 0, stream>>>(B2x, B2c, A3x, A3c, B, 8, H3, W3);
    nconv_wmma<25,5,5,2,7><<<g3, 32, 0, stream>>>(A3x, A3c, 8, 0, nullptr, nullptr, 0, 0,
                                                  f2, s2, b2, H3, W3, B3x, B3c);

    // -- decoder (concat + nearest-up2 folded into loader) -------------------
    // x34_ds = nconv4(concat(x3_ds, up2(x4_ds)))
    nconv_wmma<9,3,3,1,5><<<g2, 32, 0, stream>>>(B2x, B2c, 8, 0, B3x, B3c, 8, 1,
                                                 f4, s4, b4, H2, W2, A2x, A2c);
    // x23_ds = nconv5(concat(x2_ds, up2(x34_ds)))
    nconv_wmma<9,3,3,1,5><<<g1, 32, 0, stream>>>(A1x, A1c, 8, 0, A2x, A2c, 8, 1,
                                                 f5, s5, b5, H1, W1, B1x, B1c);
    // xout = nconv6(concat(up2(x23_ds), x1))   (upsampled half FIRST here)
    nconv_wmma<9,3,3,1,5><<<g0, 32, 0, stream>>>(B1x, B1c, 8, 1, Px, Pc, 8, 0,
                                                 f6, s6, b6, H, W, Qx, Qc);

    // -- final 1x1 ----------------------------------------------------------
    float* xout = (float*)d_out;
    float* cout = xout + (long)B * H * W;
    long npix = (long)B * H * W;
    nconv1x1<<<(unsigned)((npix + 255) / 256), 256, 0, stream>>>(
        Qx, Qc, w7sp, b7, s7, xout, cout, B, H, W);
}
